// margin_44985487458969
// MI455X (gfx1250) — matrix operand order repaired and checked by reading, not compiled, not run on hardware
//
#include <hip/hip_runtime.h>

typedef __attribute__((ext_vector_type(16))) _Float16 v16h;
typedef __attribute__((ext_vector_type(8)))  _Float16 v8h;
typedef __attribute__((ext_vector_type(8)))  float    v8f;

#define NPTS   4096
#define DFEAT  1024
#define NCLUS  8
#define TOPK   5
#define NPAIRS 28

// ---- workspace layout (bytes) ----
#define OFF_D     ((size_t)0)                                   // 64 MB fp32 distance matrix
#define OFF_XHI   (OFF_D    + (size_t)NPTS*NPTS*4)              // 8 MB f16 hi
#define OFF_XLO   (OFF_XHI  + (size_t)NPTS*DFEAT*2)             // 8 MB f16 lo
#define OFF_SQ    (OFF_XLO  + (size_t)NPTS*DFEAT*2)             // 16 KB row norms
#define OFF_CNT   (OFF_SQ   + (size_t)NPTS*4)                   // cluster counts
#define OFF_LIST  (OFF_CNT  + 256)                              // 8*4096 ints
#define OFF_SLOT  (OFF_LIST + (size_t)NCLUS*NPTS*4)             // 28 u64 min slots
#define OFF_RR    (OFF_SLOT + (size_t)NPAIRS*8)                 // removed rows
#define OFF_RC    (OFF_RR   + (size_t)NPAIRS*TOPK*4)            // removed cols
#define OFF_ACC   (OFF_RC   + (size_t)NPAIRS*TOPK*4)            // per-pair 1/min sums

__global__ void init_ws(int* counts, unsigned long long* slot, float* acc) {
    int t = threadIdx.x;
    if (t < NCLUS)  counts[t] = 0;
    if (t < NPAIRS) { slot[t] = ~0ull; acc[t] = 0.0f; }
}

// Row norms + fp32 -> (f16 hi, f16 lo) split. One wave32 per row.
__global__ void preproc(const float* __restrict__ x, _Float16* __restrict__ xhi,
                        _Float16* __restrict__ xlo, float* __restrict__ sq) {
    int wave = threadIdx.x >> 5;
    int lane = threadIdx.x & 31;
    int row  = blockIdx.x * 8 + wave;
    const float* xr = x + (size_t)row * DFEAT;
    _Float16* hr = xhi + (size_t)row * DFEAT;
    _Float16* lr = xlo + (size_t)row * DFEAT;
    float s = 0.0f;
    for (int j = lane; j < DFEAT; j += 32) {
        float v = xr[j];
        s += v * v;
        _Float16 h = (_Float16)v;
        hr[j] = h;
        lr[j] = (_Float16)(v - (float)h);
    }
    #pragma unroll
    for (int m = 16; m >= 1; m >>= 1) s += __shfl_xor(s, m, 32);
    if (lane == 0) sq[row] = s;
}

__global__ void membership(const int* __restrict__ clustering, int* counts, int* lists) {
    int i = blockIdx.x * blockDim.x + threadIdx.x;
    if (i < NPTS) {
        int c = clustering[i];
        int pos = atomicAdd(&counts[c], 1);
        lists[c * NPTS + pos] = i;
    }
}

// D[r,c] = sq[r] + sq[c] - 2 * (x_r . x_c), dot via split-f16 WMMA:
// dot ~= hi.hi + hi.lo + lo.hi  (f32 accumulate; lo.lo term ~1e-7 rel, dropped)
__global__ __launch_bounds__(256) void gemm_dist(const _Float16* __restrict__ xhi,
                                                 const _Float16* __restrict__ xlo,
                                                 const float* __restrict__ sq,
                                                 float* __restrict__ D) {
    const int lane  = threadIdx.x & 31;
    const int wave  = threadIdx.x >> 5;
    const int wm    = wave & 3, wn = wave >> 2;
    const int tile_m = blockIdx.x * 128 + wm * 32;
    const int tile_n = blockIdx.y * 64  + wn * 32;
    const int lhalf  = lane >> 4;    // 0..1
    const int l15    = lane & 15;

    v8f acc[2][2] = {};

    // A-operand (16x32 f16): lane half selects K-range 0..15 / 16..31; chunks at +0 and +8.
    const int arow0 = tile_m + l15;
    const int akb   = lhalf * 16;
    // B-operand (32x16 f16): lane half selects K-range 0..15 / 16..31, contiguous.
    const int brow0 = tile_n + l15;
    const int bkb   = lhalf * 16;

    for (int k = 0; k < DFEAT; k += 32) {
        v16h ahi[2], alo[2], bhi[2], blo[2];
        #pragma unroll
        for (int im = 0; im < 2; im++) {
            size_t base = (size_t)(arow0 + 16 * im) * DFEAT + k + akb;
            v8h h0 = *(const v8h*)(xhi + base);
            v8h h1 = *(const v8h*)(xhi + base + 8);
            ahi[im] = __builtin_shufflevector(h0, h1, 0,1,2,3,4,5,6,7,8,9,10,11,12,13,14,15);
            v8h l0 = *(const v8h*)(xlo + base);
            v8h l1 = *(const v8h*)(xlo + base + 8);
            alo[im] = __builtin_shufflevector(l0, l1, 0,1,2,3,4,5,6,7,8,9,10,11,12,13,14,15);
        }
        #pragma unroll
        for (int in = 0; in < 2; in++) {
            size_t base = (size_t)(brow0 + 16 * in) * DFEAT + k + bkb;
            bhi[in] = *(const v16h*)(xhi + base);
            blo[in] = *(const v16h*)(xlo + base);
        }
        #pragma unroll
        for (int im = 0; im < 2; im++)
            #pragma unroll
            for (int in = 0; in < 2; in++) {
                acc[im][in] = __builtin_amdgcn_wmma_f32_16x16x32_f16(
                    false, ahi[im], false, bhi[in], (short)0, acc[im][in], false, false);
                acc[im][in] = __builtin_amdgcn_wmma_f32_16x16x32_f16(
                    false, ahi[im], false, blo[in], (short)0, acc[im][in], false, false);
                acc[im][in] = __builtin_amdgcn_wmma_f32_16x16x32_f16(
                    false, alo[im], false, bhi[in], (short)0, acc[im][in], false, false);
            }
    }

    // C/D layout: VGPR v -> row = v + 8*lhalf, col = l15.
    #pragma unroll
    for (int im = 0; im < 2; im++)
        #pragma unroll
        for (int in = 0; in < 2; in++) {
            int col = tile_n + in * 16 + l15;
            float sqc = sq[col];
            #pragma unroll
            for (int v = 0; v < 8; v++) {
                int row = tile_m + im * 16 + v + 8 * lhalf;
                float d = sq[row] + sqc - 2.0f * acc[im][in][v];
                D[(size_t)row * NPTS + col] = d;
            }
        }
}

// One greedy round: block-strided min over (rows of cluster b) x (cols of cluster a),
// skipping rows/cols removed in earlier rounds; u64 (value_bits<<32 | r<<12 | c) atomicMin.
__global__ void pair_min(const float* __restrict__ D, const int* __restrict__ counts,
                         const int* __restrict__ lists, const int* __restrict__ remR,
                         const int* __restrict__ remC, int round,
                         unsigned long long* __restrict__ slot) {
    int p = blockIdx.x;
    int a = 0, b = 1;
    { int cnt = 0;
      for (int i = 0; i < NCLUS; i++)
          for (int j = i + 1; j < NCLUS; j++) { if (cnt == p) { a = i; b = j; } cnt++; } }

    const int* rowsL = lists + b * NPTS;
    const int* colsL = lists + a * NPTS;
    int nb = counts[b], na = counts[a];
    long total = (long)nb * (long)na;

    int rr[4], rc[4];
    for (int q = 0; q < round; q++) { rr[q] = remR[p * TOPK + q]; rc[q] = remC[p * TOPK + q]; }

    unsigned long long best = ~0ull;
    long stride = (long)gridDim.y * (long)blockDim.x;
    for (long e = (long)blockIdx.y * blockDim.x + threadIdx.x; e < total; e += stride) {
        int ri = (int)(e / na), ci = (int)(e % na);
        int r = rowsL[ri], c = colsL[ci];
        bool skip = false;
        for (int q = 0; q < round; q++) skip |= (r == rr[q]) | (c == rc[q]);
        if (skip) continue;
        float v = D[(size_t)r * NPTS + c];
        unsigned long long ev =
            ((unsigned long long)__float_as_uint(v) << 32) |
            (unsigned int)((r << 12) | c);
        best = (ev < best) ? ev : best;
    }

    __shared__ unsigned long long red[256];
    red[threadIdx.x] = best;
    __syncthreads();
    for (int s = 128; s > 0; s >>= 1) {
        if (threadIdx.x < s) {
            unsigned long long o = red[threadIdx.x + s];
            if (o < red[threadIdx.x]) red[threadIdx.x] = o;
        }
        __syncthreads();
    }
    if (threadIdx.x == 0) atomicMin(&slot[p], red[0]);
}

__global__ void pair_update(unsigned long long* slot, int* remR, int* remC,
                            float* acc, int round) {
    int p = threadIdx.x;
    if (p >= NPAIRS) return;
    unsigned long long m = slot[p];
    if (m != ~0ull) {
        float v = __uint_as_float((unsigned int)(m >> 32));
        int r = (int)((m >> 12) & 0xFFF);
        int c = (int)(m & 0xFFF);
        remR[p * TOPK + round] = r;
        remC[p * TOPK + round] = c;
        acc[p] += 1.0f / v;
    } else {
        remR[p * TOPK + round] = -1;
        remC[p * TOPK + round] = -1;
    }
    slot[p] = ~0ull;
}

__global__ void finalize(const float* __restrict__ acc, float* __restrict__ out) {
    float s = 0.0f;
    for (int p = 0; p < NPAIRS; p++) s += acc[p];
    out[0] = s / (float)(NCLUS * TOPK);
}

extern "C" void kernel_launch(void* const* d_in, const int* in_sizes, int n_in,
                              void* d_out, int out_size, void* d_ws, size_t ws_size,
                              hipStream_t stream) {
    const float* x          = (const float*)d_in[0];
    const int*   clustering = (const int*)d_in[1];
    float*       out        = (float*)d_out;

    char* ws = (char*)d_ws;
    float*              D      = (float*)(ws + OFF_D);
    _Float16*           xhi    = (_Float16*)(ws + OFF_XHI);
    _Float16*           xlo    = (_Float16*)(ws + OFF_XLO);
    float*              sq     = (float*)(ws + OFF_SQ);
    int*                counts = (int*)(ws + OFF_CNT);
    int*                lists  = (int*)(ws + OFF_LIST);
    unsigned long long* slot   = (unsigned long long*)(ws + OFF_SLOT);
    int*                remR   = (int*)(ws + OFF_RR);
    int*                remC   = (int*)(ws + OFF_RC);
    float*              acc    = (float*)(ws + OFF_ACC);

    init_ws<<<1, 64, 0, stream>>>(counts, slot, acc);
    preproc<<<NPTS / 8, 256, 0, stream>>>(x, xhi, xlo, sq);
    membership<<<NPTS / 256, 256, 0, stream>>>(clustering, counts, lists);
    gemm_dist<<<dim3(NPTS / 128, NPTS / 64), 256, 0, stream>>>(xhi, xlo, sq, D);
    for (int round = 0; round < TOPK; round++) {
        pair_min<<<dim3(NPAIRS, 8), 256, 0, stream>>>(D, counts, lists, remR, remC, round, slot);
        pair_update<<<1, 32, 0, stream>>>(slot, remR, remC, acc, round);
    }
    finalize<<<1, 1, 0, stream>>>(acc, out);
}
